// DSAttention_57114475102264
// MI455X (gfx1250) — matrix operand order repaired and checked by reading, not compiled, not run on hardware
//
#include <hip/hip_runtime.h>

typedef _Float16 half_t;
typedef _Float16 v16h __attribute__((ext_vector_type(16)));
typedef _Float16 v8h  __attribute__((ext_vector_type(8)));
typedef _Float16 v4h  __attribute__((ext_vector_type(4)));
typedef float    v8f  __attribute__((ext_vector_type(8)));
typedef float    v4f  __attribute__((ext_vector_type(4)));

#define WB 4
#define WL 2048
#define WH 8
#define WE 64
#define WS 2048
#define ROWSTRIDE (WH * WE)            // 512 floats between consecutive l (or s) rows
#define NELEM ((size_t)WB * WH * WL * WE)

static __device__ __forceinline__ v8f wmma_f16(v16h a, v16h b, v8f c) {
  return __builtin_amdgcn_wmma_f32_16x16x32_f16(false, a, false, b, (short)0, c, false, false);
}

// ---- f32 fallback-path helpers (round-1 layouts) -------------------------

static __device__ __forceinline__ v16h f32x16_to_h16(const float* __restrict__ p) {
  v4f f0 = *(const v4f*)(p);
  v4f f1 = *(const v4f*)(p + 4);
  v4f f2 = *(const v4f*)(p + 8);
  v4f f3 = *(const v4f*)(p + 12);
  v16h r;
#pragma unroll
  for (int i = 0; i < 4; ++i) {
    r[i]      = (half_t)f0[i];
    r[i + 4]  = (half_t)f1[i];
    r[i + 8]  = (half_t)f2[i];
    r[i + 12] = (half_t)f3[i];
  }
  return r;
}

// Q A-fragment (16x32 f16): M = lane%16; elems 0..15 -> K = 16*half1 + i.
static __device__ __forceinline__ v16h load_q_afrag(const float* __restrict__ qrow,
                                                    int c, int half1) {
  const float* g0 = qrow + 32 * c + 16 * half1;
  const float* g1 = g0 + 8;
  v4f f0 = *(const v4f*)(g0);
  v4f f1 = *(const v4f*)(g0 + 4);
  v4f f2 = *(const v4f*)(g1);
  v4f f3 = *(const v4f*)(g1 + 4);
  v16h r;
#pragma unroll
  for (int i = 0; i < 4; ++i) {
    r[i]      = (half_t)f0[i];
    r[i + 4]  = (half_t)f1[i];
    r[i + 8]  = (half_t)f2[i];
    r[i + 12] = (half_t)f3[i];
  }
  return r;
}

static __device__ __forceinline__ void wave_lds_fence() {
  __builtin_amdgcn_wave_barrier();
#if __has_builtin(__builtin_amdgcn_s_wait_dscnt)
  __builtin_amdgcn_s_wait_dscnt(0);
#else
  asm volatile("s_wait_dscnt 0" ::: "memory");
#endif
  __builtin_amdgcn_wave_barrier();
}

// ======================= prologue kernels (PRE path) =======================

// kh[b][h][s][e] (f16) <- k[b][s][h][e] (f32). One thread per 4 elements.
__global__ __launch_bounds__(256)
void cvt_k_kernel(const float* __restrict__ k, half_t* __restrict__ kh) {
  const int idx4 = blockIdx.x * 256 + threadIdx.x;     // < 2^20
  const int e4 = idx4 & 15;
  const int h  = (idx4 >> 4) & (WH - 1);
  const int s  = (idx4 >> 7) & (WL - 1);
  const int b  = idx4 >> 18;
  v4f f = *(const v4f*)(k + ((size_t)((b * WL + s) * WH + h)) * WE + e4 * 4);
  v4h o;
#pragma unroll
  for (int i = 0; i < 4; ++i) o[i] = (half_t)f[i];
  *(v4h*)(kh + ((size_t)((b * WH + h) * WL + s)) * WE + e4 * 4) = o;
}

// vt[b][h][d][s] (f16) <- v[b][s][h][d] (f32). LDS-tiled 64x64 transpose.
__global__ __launch_bounds__(256)
void transpose_v_kernel(const float* __restrict__ v, half_t* __restrict__ vt) {
  __shared__ half_t tile[64][72];                      // padded vs bank conflicts
  const int bh = blockIdx.x >> 5;                      // 32 (b,h) pairs
  const int s0 = (blockIdx.x & 31) * 64;               // 32 s-tiles of 64
  const int b = bh >> 3, h = bh & (WH - 1);
  const int t = threadIdx.x;

  {                                                    // load 64 s-rows x 64 d, coalesced
    const int row = t >> 2, cb = (t & 3) * 16;
    const float* src = v + ((size_t)((b * WL + s0 + row) * WH + h)) * WE + cb;
#pragma unroll
    for (int g = 0; g < 4; ++g) {
      v4f f = *(const v4f*)(src + g * 4);
#pragma unroll
      for (int i = 0; i < 4; ++i) tile[row][cb + g * 4 + i] = (half_t)f[i];
    }
  }
  __syncthreads();
  {                                                    // store d-major, coalesced 32B chunks
    const int d = t >> 2, sb = (t & 3) * 16;
    v16h o;
#pragma unroll
    for (int j = 0; j < 16; ++j) o[j] = tile[sb + j][d];
    *(v16h*)(vt + ((size_t)(bh * WE + d)) * WL + s0 + sb) = o;
  }
}

// ============================ main kernel ============================

template <bool PRE>
__global__ __launch_bounds__(256)
void dsattn_fwd_kernel(const float* __restrict__ q, const float* __restrict__ k,
                       const float* __restrict__ v, const float* __restrict__ tau,
                       const float* __restrict__ delta,
                       const half_t* __restrict__ kh, const half_t* __restrict__ vt,
                       float* __restrict__ out, float* __restrict__ attn) {
  __shared__ half_t stage[8][16 * 32];   // 1 KB per wave: 16 rows x 32 keys, f16

  const int tid   = threadIdx.x;
  const int w     = tid >> 5;
  const int lane  = tid & 31;
  const int lid   = lane & 15;
  const int half1 = lane >> 4;

  const int wid   = blockIdx.x * 8 + w;  // 4096 waves total
  const int ltile = wid & 127;
  const int bh    = wid >> 7;
  const int h     = bh & (WH - 1);
  const int b     = bh >> 3;
  const int l0    = ltile * 16;

  const float scale = 0.125f;            // 1/sqrt(64)
  const float tsc   = tau[b] * scale;

  // ---- Q A-fragments (two K=32 chunks over E=64) ----
  const float* qrow = q + ((size_t)((b * WL + l0 + lid) * WH + h)) * WE;
  const v16h qf0 = load_q_afrag(qrow, 0, half1);
  const v16h qf1 = load_q_afrag(qrow, 1, half1);

  // per-lane bases
  const float*  kcol  = k  + ((size_t)((b * WL + lid) * WH + h)) * WE + 16 * half1;
  const half_t* khcol = kh + ((size_t)(bh * WL + lid)) * WE + 16 * half1;
  const float*  vcol  = v  + ((size_t)((b * WL + 16 * half1) * WH + h)) * WE + lid;
  const half_t* vtcol = vt + ((size_t)(bh * WE + lid)) * WL + 16 * half1;
  const float*  dbase = delta + b * WL + lid;

  // one 16x16 score tile at key offset s0 (f32 accum)
  auto qk = [&](int s0) -> v8f {
    v8f c = {0.f, 0.f, 0.f, 0.f, 0.f, 0.f, 0.f, 0.f};
    if constexpr (PRE) {
      const half_t* p = khcol + (size_t)s0 * WE;
      v16h b0 = *(const v16h*)(p);
      v16h b1 = *(const v16h*)(p + 32);
      c = wmma_f16(qf0, b0, c);
      c = wmma_f16(qf1, b1, c);
    } else {
      const float* p = kcol + (size_t)s0 * ROWSTRIDE;
      v16h b0 = f32x16_to_h16(p);
      v16h b1 = f32x16_to_h16(p + 32);
      c = wmma_f16(qf0, b0, c);
      c = wmma_f16(qf1, b1, c);
    }
    return c;
  };

  // ================= pass 1: exact row max =================
  v8f vmax;
#pragma unroll
  for (int i = 0; i < 8; ++i) vmax[i] = -3.0e38f;

  for (int st = 0; st < 128; ++st) {
    const int s0 = st * 16;
    v8f c = qk(s0);
    const float dsc = dbase[s0] * scale;
#pragma unroll
    for (int i = 0; i < 8; ++i) vmax[i] = fmaxf(vmax[i], c[i] * tsc + dsc);
  }
#pragma unroll
  for (int off = 8; off; off >>= 1) {
#pragma unroll
    for (int i = 0; i < 8; ++i) vmax[i] = fmaxf(vmax[i], __shfl_xor(vmax[i], off, 32));
  }

  // ========== pass 2: row sums + out = (exp tiles) @ V via WMMA ==========
  v8f acc[4] = {};
  v8f vsum = {};
  half_t* st_row = &stage[w][0];

  for (int sc = 0; sc < 64; ++sc) {      // 32 keys / iteration
    const int sbase = sc * 32;
#pragma unroll
    for (int t = 0; t < 2; ++t) {
      const int s0 = sbase + t * 16;
      v8f c = qk(s0);
      const float dsc = dbase[s0] * scale;
#pragma unroll
      for (int i = 0; i < 8; ++i) {
        float e = __expf(c[i] * tsc + dsc - vmax[i]);
        vsum[i] += e;
        st_row[(i + 8 * half1) * 32 + t * 16 + lid] = (half_t)e;   // C -> row-major
      }
    }
    wave_lds_fence();

    // attention tile as A-fragment (16x32 f16), M = lid
    const half_t* erow = st_row + lid * 32;
    v8h lo = *(const v8h*)(erow + 16 * half1);
    v8h hi = *(const v8h*)(erow + 16 * half1 + 8);
    v16h ef = __builtin_shufflevector(lo, hi, 0, 1, 2, 3, 4, 5, 6, 7,
                                      8, 9, 10, 11, 12, 13, 14, 15);

#pragma unroll
    for (int ds = 0; ds < 4; ++ds) {
      v16h vf;
      if constexpr (PRE) {
        vf = *(const v16h*)(vtcol + (size_t)ds * 16 * WL + sbase);
      } else {
        const float* vp = vcol + (size_t)sbase * ROWSTRIDE + ds * 16;
#pragma unroll
        for (int i = 0; i < 16; ++i) vf[i] = (half_t)vp[(size_t)i * ROWSTRIDE];
      }
      acc[ds] = wmma_f16(ef, vf, acc[ds]);
    }
    wave_lds_fence();                    // WAR vs next iteration's LDS stores
  }

#pragma unroll
  for (int off = 8; off; off >>= 1) {
#pragma unroll
    for (int i = 0; i < 8; ++i) vsum[i] += __shfl_xor(vsum[i], off, 32);
  }
  v8f rinv;
#pragma unroll
  for (int i = 0; i < 8; ++i) rinv[i] = 1.0f / vsum[i];

  // out[b, l0+m, h, d]
  float* orow = out + ((size_t)((b * WL + l0 + 8 * half1) * WH + h)) * WE + lid;
#pragma unroll
  for (int ds = 0; ds < 4; ++ds) {
#pragma unroll
    for (int i = 0; i < 8; ++i)
      orow[(size_t)i * ROWSTRIDE + ds * 16] = acc[ds][i] * rinv[i];
  }

  // ===== pass 3: recompute, stream normalized a[b,h,l,s] (non-temporal) =====
  float* arow = attn + ((size_t)((b * WH + h) * WL + l0 + 8 * half1)) * WS + lid;
  for (int st = 0; st < 128; ++st) {
    const int s0 = st * 16;
    v8f c = qk(s0);
    const float dsc = dbase[s0] * scale;
#pragma unroll
    for (int i = 0; i < 8; ++i) {
      float aval = __expf(c[i] * tsc + dsc - vmax[i]) * rinv[i];
      __builtin_nontemporal_store(aval, arow + (size_t)i * WS + s0);
    }
  }
}

extern "C" void kernel_launch(void* const* d_in, const int* in_sizes, int n_in,
                              void* d_out, int out_size, void* d_ws, size_t ws_size,
                              hipStream_t stream) {
  const float* q     = (const float*)d_in[0];
  const float* k     = (const float*)d_in[1];
  const float* v     = (const float*)d_in[2];
  const float* tau   = (const float*)d_in[3];
  const float* delta = (const float*)d_in[4];

  float* out  = (float*)d_out;
  float* attn = out + (size_t)WB * WL * WH * WE;       // tuple: (out, a)

  const size_t need = 2 * NELEM * sizeof(half_t);      // kh + vt = 16 MB
  if (ws_size >= need) {
    half_t* kh = (half_t*)d_ws;
    half_t* vt = kh + NELEM;
    cvt_k_kernel<<<(int)(NELEM / 4 / 256), 256, 0, stream>>>(k, kh);
    transpose_v_kernel<<<WB * WH * (WL / 64), 256, 0, stream>>>(v, vt);
    dsattn_fwd_kernel<true><<<512, 256, 0, stream>>>(q, k, v, tau, delta, kh, vt, out, attn);
  } else {
    dsattn_fwd_kernel<false><<<512, 256, 0, stream>>>(q, k, v, tau, delta,
                                                      (const half_t*)nullptr,
                                                      (const half_t*)nullptr, out, attn);
  }
}
